// Mamba2Block_36361193128110
// MI455X (gfx1250) — matrix operand order repaired and checked by reading, not compiled, not run on hardware
//
#include <hip/hip_runtime.h>
#include <hip/hip_bf16.h>
#include <math.h>

// ---------------------------------------------------------------------------
// Mamba-2 block forward for MI455X (gfx1250, wave32, WMMA).
// Precision: bf16 WMMA operands, f32 accumulate (compute-bound workload,
// AI ~650 FLOP/B, so we ride the 16x16x32 bf16 matrix pipe).
// ---------------------------------------------------------------------------

typedef __bf16 bf16;
typedef __attribute__((ext_vector_type(16))) __bf16 v16bf;
typedef __attribute__((ext_vector_type(8)))  __bf16 v8bf;
typedef __attribute__((ext_vector_type(8)))  float  v8f;

#define DIM_      2048
#define L_        2048
#define BATCH_    2
#define D_INNER_  4096
#define NHEADS_   64
#define HEADDIM_  64
#define D_STATE_  64
#define D_CONV_   4
#define CHUNK_    64
#define NCHUNK_   (L_ / CHUNK_)
#define PROJ_     (2 * D_INNER_ + 2 * D_STATE_ + NHEADS_)   /* 8384 */
#define COL_Z_    (D_INNER_)                                 /* 4096 */
#define COL_B_    (2 * D_INNER_)                             /* 8192 */
#define COL_C_    (2 * D_INNER_ + D_STATE_)                  /* 8256 */
#define COL_DT_   (2 * D_INNER_ + 2 * D_STATE_)              /* 8320 */

// ---- WMMA fragment loaders (ISA 7.12.2 VGPR layouts, wave32) --------------
// A 16x32 bf16: lanes 0-15 hold row M=lane, K {0..7, 16..23};
//               lanes 16-31 hold row M=lane-16, K {8..15, 24..31}.
__device__ __forceinline__ v16bf load_a_frag(const bf16* base, int ld, int lane) {
  const bf16* p = base + (size_t)(lane & 15) * ld + ((lane >> 4) << 3);
  union { v16bf v; v8bf h[2]; } u;
  u.h[0] = *(const v8bf*)(p);
  u.h[1] = *(const v8bf*)(p + 16);
  return u.v;
}
// B 32x16 bf16 as (N,K) rows: lane n holds column N=n; lanes 0-15 K {0..7, 16..23},
// lanes 16-31 K {8..15, 24..31} (same K order as A).
__device__ __forceinline__ v16bf load_b_frag(const bf16* base, int ld, int lane) {
  const bf16* p = base + (size_t)(lane & 15) * ld + ((lane >> 4) << 3);
  union { v16bf v; v8bf h[2]; } u;
  u.h[0] = *(const v8bf*)(p);
  u.h[1] = *(const v8bf*)(p + 16);
  return u.v;
}

// ---------------------------------------------------------------------------
// Kernel 1/5: C[M,N] = A[M,K] * W[N,K]^T   (A row-major f32, W (N,K) f32)
// 128x128x32 tile, 256 threads = 8 waves (2x4), bf16 staging in LDS.
// ---------------------------------------------------------------------------
__global__ __launch_bounds__(256)
void wmma_gemm_nt(const float* __restrict__ A, const float* __restrict__ W,
                  float* __restrict__ C, int M, int N, int K) {
  __shared__ __attribute__((aligned(32))) bf16 As[128][32];
  __shared__ __attribute__((aligned(32))) bf16 Bs[128][32];
  const int tid  = threadIdx.x;
  const int lane = tid & 31, wave = tid >> 5;
  const int wm = wave >> 2, wn = wave & 3;        // 2 x 4 wave grid
  const int m0 = blockIdx.y * 128, n0 = blockIdx.x * 128;
  const int r  = tid >> 1, cb = (tid & 1) << 4;   // staging: 16 floats/thread
  const int nr = n0 + r;

  v8f acc[4][2] = {};                             // 64x32 per wave

  for (int k0 = 0; k0 < K; k0 += 32) {
    const float* pa = A + (size_t)(m0 + r) * K + k0 + cb;
    #pragma unroll
    for (int i = 0; i < 16; ++i) As[r][cb + i] = (bf16)pa[i];
    if (nr < N) {
      const float* pb = W + (size_t)nr * K + k0 + cb;
      #pragma unroll
      for (int i = 0; i < 16; ++i) Bs[r][cb + i] = (bf16)pb[i];
      if (k0 + 32 < K) __builtin_prefetch(pb + 32, 0, 1);   // global_prefetch_b8
    } else {
      #pragma unroll
      for (int i = 0; i < 16; ++i) Bs[r][cb + i] = (bf16)0.f;
    }
    if (k0 + 32 < K) __builtin_prefetch(pa + 32, 0, 1);
    __syncthreads();

    #pragma unroll
    for (int mi = 0; mi < 4; ++mi) {
      v16bf af = load_a_frag(&As[wm * 64 + mi * 16][0], 32, lane);
      #pragma unroll
      for (int ni = 0; ni < 2; ++ni) {
        v16bf bf_ = load_b_frag(&Bs[wn * 32 + ni * 16][0], 32, lane);
        acc[mi][ni] = __builtin_amdgcn_wmma_f32_16x16x32_bf16(
            false, af, false, bf_, (short)0, acc[mi][ni], false, false);
      }
    }
    __syncthreads();
  }

  // D layout: n = lane&15, m = 8*(lane>>4) + vgpr
  const int cn = lane & 15, half = lane >> 4;
  #pragma unroll
  for (int mi = 0; mi < 4; ++mi)
    #pragma unroll
    for (int ni = 0; ni < 2; ++ni) {
      int gn = n0 + wn * 32 + ni * 16 + cn;
      if (gn < N) {
        int gm = m0 + wm * 64 + mi * 16 + half * 8;
        #pragma unroll
        for (int rr = 0; rr < 8; ++rr)
          C[(size_t)(gm + rr) * N + gn] = acc[mi][ni][rr];
      }
    }
}

// ---------------------------------------------------------------------------
// Kernel 2: depthwise conv(4) + SiLU + dt softplus + discretization.
// ---------------------------------------------------------------------------
__global__ __launch_bounds__(256)
void conv_disc_kernel(const float* __restrict__ proj, const float* __restrict__ conv_w,
                      const float* __restrict__ conv_b, const float* __restrict__ A_log,
                      const float* __restrict__ dt_bias,
                      float* __restrict__ xact, float* __restrict__ Xd,
                      float* __restrict__ Ad) {
  size_t idx = (size_t)blockIdx.x * blockDim.x + threadIdx.x;
  if (idx >= (size_t)BATCH_ * L_ * D_INNER_) return;
  const int    ch = (int)(idx & (D_INNER_ - 1));
  const size_t bl = idx >> 12;                 // row (b*L + l)
  const int    l  = (int)(bl & (L_ - 1));

  float a = conv_b[ch];
  #pragma unroll
  for (int k = 0; k < D_CONV_; ++k) {
    int ll = l + k - (D_CONV_ - 1);
    if (ll >= 0) {
      size_t rr = bl - (size_t)(D_CONV_ - 1 - k);
      a = fmaf(conv_w[ch * D_CONV_ + k], proj[rr * PROJ_ + ch], a);
    }
  }
  float xa = a / (1.f + __expf(-a));           // SiLU
  xact[idx] = xa;

  const int h = ch >> 6;                       // / HEADDIM
  float dtr = proj[bl * PROJ_ + COL_DT_ + h] + dt_bias[h];
  float dtf = (dtr > 20.f) ? dtr : log1pf(__expf(dtr));   // softplus
  Xd[idx] = xa * dtf;                          // X_disc
  if ((ch & (HEADDIM_ - 1)) == 0)
    Ad[bl * NHEADS_ + h] = -__expf(A_log[h]) * dtf;       // A_disc
}

// ---------------------------------------------------------------------------
// Kernel 3: chunkwise SSD scan. One workgroup per (b, head); 4 waves.
// Carried 64x64 state lives in wave registers and re-enters WMMA as the
// C operand -> the 32-chunk recurrence is fused in a single kernel.
// ---------------------------------------------------------------------------
__global__ __launch_bounds__(128)
void ssd_scan_kernel(const float* __restrict__ proj, const float* __restrict__ Xd,
                     const float* __restrict__ Ad, float* __restrict__ Y) {
  __shared__ __attribute__((aligned(32))) bf16 Cl [64][64];  // C  (l,n)
  __shared__ __attribute__((aligned(32))) bf16 Bl [64][64];  // B  (s,n)
  __shared__ __attribute__((aligned(32))) bf16 Xt [64][64];  // Xᵀ (p,l)
  __shared__ __attribute__((aligned(32))) bf16 Bdt[64][64];  // (B∘decay)ᵀ (n,l)
  __shared__ __attribute__((aligned(32))) bf16 Gl [64][64];  // masked G (l,s)
  __shared__ __attribute__((aligned(32))) bf16 Sl [64][64];  // state (p,n) bf16
  __shared__ float Acum[64];
  __shared__ float sE;

  const int bh = blockIdx.x;
  const int b = bh / NHEADS_, h = bh % NHEADS_;
  const int tid = threadIdx.x, lane = tid & 31, wave = tid >> 5;
  const int cn = lane & 15, half = lane >> 4;

  v8f S[4] = {};   // state rows p = 16*wave + 8*half + r ; cols n = 16*nt + cn

  for (int c = 0; c < NCHUNK_; ++c) {
    __syncthreads();                           // prev chunk fully consumed
    const size_t row0 = (size_t)b * L_ + (size_t)c * CHUNK_;

    if (tid == 0) {                            // tiny serial cumsum of A
      float run = 0.f;
      for (int i = 0; i < CHUNK_; ++i) {
        run += Ad[(row0 + i) * NHEADS_ + h];
        Acum[i] = run;
      }
      sE = run;
    }
    __syncthreads();
    const float E = sE;

    // ---- stage chunk operands (fp32 -> bf16), transposing X and B∘decay ----
    for (int t = tid; t < 64 * 64; t += 128) {
      int l = t >> 6, n = t & 63;              // n doubles as p for X
      float bv = proj[(row0 + l) * PROJ_ + COL_B_ + n];
      float cv = proj[(row0 + l) * PROJ_ + COL_C_ + n];
      float xv = Xd[(row0 + l) * D_INNER_ + h * HEADDIM_ + n];
      Bl[l][n]  = (bf16)bv;
      Cl[l][n]  = (bf16)cv;
      Xt[n][l]  = (bf16)xv;
      Bdt[n][l] = (bf16)(bv * __expf(E - Acum[l]));       // decay_states
    }
    // stage state entering this chunk as bf16 (p,n)
    #pragma unroll
    for (int nt = 0; nt < 4; ++nt)
      #pragma unroll
      for (int rr = 0; rr < 8; ++rr)
        Sl[wave * 16 + half * 8 + rr][nt * 16 + cn] = (bf16)S[nt][rr];
    __syncthreads();

    // ---- G = C·Bᵀ (contract n), then mask∘scale with exp(segsum A) --------
    #pragma unroll
    for (int st = 0; st < 4; ++st) {
      v8f g = {};
      #pragma unroll
      for (int kk = 0; kk < 2; ++kk) {
        v16bf af  = load_a_frag(&Cl[wave * 16][kk * 32], 64, lane);
        v16bf bf_ = load_b_frag(&Bl[st * 16][kk * 32], 64, lane);
        g = __builtin_amdgcn_wmma_f32_16x16x32_bf16(false, af, false, bf_,
                                                    (short)0, g, false, false);
      }
      #pragma unroll
      for (int rr = 0; rr < 8; ++rr) {
        int gl = wave * 16 + half * 8 + rr;
        int gs = st * 16 + cn;
        float v = (gl >= gs) ? g[rr] * __expf(Acum[gl] - Acum[gs]) : 0.f;
        Gl[gl][gs] = (bf16)v;
      }
    }
    __syncthreads();

    // ---- Y = exp(Acum)∘(C·Sᵀ) + (mask∘G)·X --------------------------------
    #pragma unroll
    for (int pt = 0; pt < 4; ++pt) {
      v8f y = {};
      #pragma unroll
      for (int kk = 0; kk < 2; ++kk) {         // Y_off: contract n
        v16bf af  = load_a_frag(&Cl[wave * 16][kk * 32], 64, lane);
        v16bf bf_ = load_b_frag(&Sl[pt * 16][kk * 32], 64, lane);
        y = __builtin_amdgcn_wmma_f32_16x16x32_bf16(false, af, false, bf_,
                                                    (short)0, y, false, false);
      }
      #pragma unroll
      for (int rr = 0; rr < 8; ++rr)
        y[rr] *= __expf(Acum[wave * 16 + half * 8 + rr]);  // state_decay_out
      #pragma unroll
      for (int kk = 0; kk < 2; ++kk) {         // Y_diag: contract s
        v16bf af  = load_a_frag(&Gl[wave * 16][kk * 32], 64, lane);
        v16bf bf_ = load_b_frag(&Xt[pt * 16][kk * 32], 64, lane);
        y = __builtin_amdgcn_wmma_f32_16x16x32_bf16(false, af, false, bf_,
                                                    (short)0, y, false, false);
      }
      #pragma unroll
      for (int rr = 0; rr < 8; ++rr) {
        int gl = wave * 16 + half * 8 + rr;
        int gp = pt * 16 + cn;
        Y[(row0 + gl) * D_INNER_ + h * HEADDIM_ + gp] = y[rr];
      }
    }

    // ---- state update: S = e^E·S + Xᵀ·(B∘decay)  (contract l) -------------
    const float eE = __expf(E);
    #pragma unroll
    for (int nt = 0; nt < 4; ++nt) {
      #pragma unroll
      for (int rr = 0; rr < 8; ++rr) S[nt][rr] *= eE;
      #pragma unroll
      for (int kk = 0; kk < 2; ++kk) {
        v16bf af  = load_a_frag(&Xt[wave * 16][kk * 32], 64, lane);
        v16bf bf_ = load_b_frag(&Bdt[nt * 16][kk * 32], 64, lane);
        S[nt] = __builtin_amdgcn_wmma_f32_16x16x32_bf16(false, af, false, bf_,
                                                        (short)0, S[nt], false, false);
      }
    }
  }
}

// ---------------------------------------------------------------------------
// Kernel 4: yg = (Y + D_skip * x_act) * silu(z)
// ---------------------------------------------------------------------------
__global__ __launch_bounds__(256)
void gate_kernel(const float* __restrict__ Yb, const float* __restrict__ xact,
                 const float* __restrict__ proj, const float* __restrict__ D_skip,
                 float* __restrict__ yg) {
  size_t idx = (size_t)blockIdx.x * blockDim.x + threadIdx.x;
  if (idx >= (size_t)BATCH_ * L_ * D_INNER_) return;
  const int    ch = (int)(idx & (D_INNER_ - 1));
  const size_t bl = idx >> 12;
  float z  = proj[bl * PROJ_ + COL_Z_ + ch];
  float sz = z / (1.f + __expf(-z));
  yg[idx]  = (Yb[idx] + D_skip[ch] * xact[idx]) * sz;
}

// ---------------------------------------------------------------------------
extern "C" void kernel_launch(void* const* d_in, const int* in_sizes, int n_in,
                              void* d_out, int out_size, void* d_ws, size_t ws_size,
                              hipStream_t stream) {
  const float* x          = (const float*)d_in[0];
  const float* in_proj_w  = (const float*)d_in[1];   // (PROJ_, DIM) row-major = (N,K)
  const float* conv_w     = (const float*)d_in[2];
  const float* conv_b     = (const float*)d_in[3];
  const float* out_proj_w = (const float*)d_in[4];   // (DIM, D_INNER) = (N,K)
  const float* A_log      = (const float*)d_in[5];
  const float* D_skip     = (const float*)d_in[6];
  const float* dt_bias    = (const float*)d_in[7];
  float* out = (float*)d_out;

  const size_t BL = (size_t)BATCH_ * L_;             // 4096
  // Workspace carve-out (~340 MB of f32):
  float* ws   = (float*)d_ws;
  float* proj = ws;                                  // BL * PROJ_
  float* xact = proj + BL * PROJ_;                   // BL * D_INNER_
  float* Xd   = xact + BL * D_INNER_;                // BL * D_INNER_
  float* Ad   = Xd   + BL * D_INNER_;                // BL * NHEADS_
  float* Yb   = Ad   + BL * NHEADS_;                 // BL * D_INNER_
  float* yg   = Xd;  // X_disc fully consumed by SSD before gating: reuse

  dim3 blk(256);

  // 1) proj = x @ in_proj_w^T : 4096 x 8384 x 2048
  dim3 g1((PROJ_ + 127) / 128, (unsigned)(BL / 128));
  wmma_gemm_nt<<<g1, blk, 0, stream>>>(x, in_proj_w, proj, (int)BL, PROJ_, DIM_);

  // 2) conv + SiLU + discretize
  const int tot = (int)(BL * D_INNER_);
  conv_disc_kernel<<<(tot + 255) / 256, 256, 0, stream>>>(
      proj, conv_w, conv_b, A_log, dt_bias, xact, Xd, Ad);

  // 3) SSD chunk scan: one block per (b, head), 4 waves, state in registers
  ssd_scan_kernel<<<BATCH_ * NHEADS_, 128, 0, stream>>>(proj, Xd, Ad, Yb);

  // 4) gating
  gate_kernel<<<(tot + 255) / 256, 256, 0, stream>>>(Yb, xact, proj, D_skip, yg);

  // 5) out = yg @ out_proj_w^T : 4096 x 2048 x 4096
  dim3 g2(DIM_ / 128, (unsigned)(BL / 128));
  wmma_gemm_nt<<<g2, blk, 0, stream>>>(yg, out_proj_w, out, (int)BL, DIM_, D_INNER_);
}
